// SpatialAttention_28424093564998
// MI455X (gfx1250) — matrix operand order repaired and checked by reading, not compiled, not run on hardware
//
#include <hip/hip_runtime.h>
#include <hip/hip_bf16.h>

typedef _Float16 v16h __attribute__((ext_vector_type(16)));
typedef _Float16 v8h  __attribute__((ext_vector_type(8)));
typedef float    v8f  __attribute__((ext_vector_type(8)));
typedef float    v4f  __attribute__((ext_vector_type(4)));

#define CAT16(lo, hi) __builtin_shufflevector((lo), (hi), 0,1,2,3,4,5,6,7,8,9,10,11,12,13,14,15)
#define WMMA_F16(A, B, C) __builtin_amdgcn_wmma_f32_16x16x32_f16(false, (A), false, (B), (short)0, (C), false, false)

constexpr int BB = 8;
constexpr int CC = 64;
constexpr int NN = 4096;

// ---------------------------------------------------------------------------
// Fast transcendental / cross-lane helpers (all VALU, no LDS)
// ---------------------------------------------------------------------------
__device__ __forceinline__ float fast_exp2(float x) {
#if __has_builtin(__builtin_amdgcn_exp2f)
    return __builtin_amdgcn_exp2f(x);     // raw v_exp_f32
#else
    return exp2f(x);
#endif
}

// DPP16 ROW_XMASK move: lane <- lane ^ mask (within its row of 16 lanes)
template <int XMASK>
__device__ __forceinline__ float dpp_xor(float x) {
    const int xi = __builtin_bit_cast(int, x);
    const int r  = __builtin_amdgcn_update_dpp(0, xi, 0x160 + XMASK, 0xF, 0xF, true);
    return __builtin_bit_cast(float, r);
}
__device__ __forceinline__ float half16_max(float m) {
    m = fmaxf(m, dpp_xor<1>(m));
    m = fmaxf(m, dpp_xor<2>(m));
    m = fmaxf(m, dpp_xor<4>(m));
    m = fmaxf(m, dpp_xor<8>(m));
    return m;                               // reduced over the 16-lane half
}
__device__ __forceinline__ float half16_sum(float s) {
    s += dpp_xor<1>(s);
    s += dpp_xor<2>(s);
    s += dpp_xor<4>(s);
    s += dpp_xor<8>(s);
    return s;
}

// ---------------------------------------------------------------------------
// Kernel 1: f32 -> f16 conversion producing both layouts:
//   xh[b][c][n]  (input layout, f16)  -> B-fragments of P@V
//   xT[b][n][c]  (transposed,  f16)   -> A-fragments + B-fragments of S
// ---------------------------------------------------------------------------
__global__ __launch_bounds__(256) void convert_f16_kernel(const float* __restrict__ x,
                                                          _Float16* __restrict__ xT,
                                                          _Float16* __restrict__ xh) {
    __shared__ __align__(16) _Float16 tile[64][72];
    const int b  = blockIdx.x >> 6;
    const int n0 = (blockIdx.x & 63) << 6;
    const int t  = threadIdx.x;
    const int nl = t & 63;
    const int c4 = t >> 6;

#pragma unroll
    for (int co = 0; co < 16; ++co) {
        const int c = (co << 2) + c4;
        const size_t gi = ((size_t)b * CC + c) * NN + n0 + nl;
        const _Float16 h = (_Float16)x[gi];
        xh[gi] = h;
        tile[nl][c] = h;
    }
    __syncthreads();

    const int row = t >> 2;
    const int q   = t & 3;
    const v8h* src = (const v8h*)&tile[row][q * 16];
    v8h* dst = (v8h*)(xT + ((size_t)b * NN + n0 + row) * CC + q * 16);
    dst[0] = src[0];
    dst[1] = src[1];
}

// ---------------------------------------------------------------------------
// Helpers
// ---------------------------------------------------------------------------
__device__ __forceinline__ v16h load_bfrag(const _Float16* p) {
    const v8h lo = *(const v8h*)p;
    const v8h hi = *(const v8h*)(p + 16);
    return CAT16(lo, hi);
}

// One 16x16 S tile (rows from A-frags, cols m0..m0+15), K = C = 64.
// A-frags are pre-scaled by log2(e), so the result is in log2 domain.
__device__ __forceinline__ v8f s_tile16(const _Float16* __restrict__ xTb, int m0,
                                        int ln, int hf, v16h a0, v16h a1) {
    const _Float16* pb = xTb + (size_t)(m0 + ln) * CC + hf * 8;
    const v16h b0 = load_bfrag(pb);
    const v16h b1 = load_bfrag(pb + 32);
    v8f acc = {};
    acc = WMMA_F16(a0, b0, acc);
    acc = WMMA_F16(a1, b1, acc);
    return acc;
}

// ---------------------------------------------------------------------------
// Kernel 2: fused attention. One wave owns a 16-row strip of one batch.
// ---------------------------------------------------------------------------
__global__ __launch_bounds__(128) void spatial_attn_kernel(const _Float16* __restrict__ xT,
                                                           const _Float16* __restrict__ xh,
                                                           float* __restrict__ out_y,
                                                           float* __restrict__ out_attn) {
    __shared__ __align__(16) _Float16 ldsP[4][16 * 40];   // per-wave 16x32 P tile, pad->40

    const int lane = threadIdx.x & 31;
    const int wid  = threadIdx.x >> 5;
    const int ln   = lane & 15;
    const int hf   = lane >> 4;
    const int gs   = blockIdx.x * 4 + wid;
    const int b    = gs >> 8;
    const int r0   = (gs & 255) << 4;

    const _Float16* xTb = xT + (size_t)b * NN * CC;
    const _Float16* xhb = xh + (size_t)b * CC * NN;

    // A fragments (persist in regs). Scale by log2(e) so S-WMMAs emit log2-domain scores.
    const _Float16* pa = xTb + (size_t)(r0 + ln) * CC + hf * 8;
    v16h a0 = CAT16(*(const v8h*)pa,        *(const v8h*)(pa + 16));
    v16h a1 = CAT16(*(const v8h*)(pa + 32), *(const v8h*)(pa + 48));
#pragma unroll
    for (int i = 0; i < 16; ++i) {
        a0[i] *= (_Float16)1.44269504088896f;
        a1[i] *= (_Float16)1.44269504088896f;
    }

    v8f yacc0 = {}, yacc1 = {}, yacc2 = {}, yacc3 = {};
    float rmax[8], psum[8];
#pragma unroll
    for (int r = 0; r < 8; ++r) { rmax[r] = -__builtin_huge_valf(); psum[r] = 0.0f; }

    _Float16* lp = &ldsP[wid][0];

    // ---------------- Pass 1: online softmax + Y accumulation ----------------
    for (int g = 0; g < NN / 32; ++g) {
        const int m0 = g << 5;
        v8f s0 = s_tile16(xTb, m0,      ln, hf, a0, a1);
        v8f s1 = s_tile16(xTb, m0 + 16, ln, hf, a0, a1);

        v8f p0, p1;
        float corr[8];
#pragma unroll
        for (int r = 0; r < 8; ++r) {                 // row = r + hf*8 in every D tile
            const float m  = half16_max(fmaxf(s0[r], s1[r]));   // DPP butterfly, VALU only
            const float nm = fmaxf(rmax[r], m);
            const float c  = fast_exp2(rmax[r] - nm);
            rmax[r] = nm;
            const float e0 = fast_exp2(s0[r] - nm);
            const float e1 = fast_exp2(s1[r] - nm);
            p0[r] = e0; p1[r] = e1;
            psum[r] = psum[r] * c + (e0 + e1);        // per-lane partial sum (no butterfly)
            corr[r] = c;
        }

        // D-layout -> A-layout reshape through wave-private LDS (f16, padded rows)
#pragma unroll
        for (int r = 0; r < 8; ++r) {
            const int row = r + hf * 8;
            lp[row * 40 + ln]      = (_Float16)p0[r];
            lp[row * 40 + 16 + ln] = (_Float16)p1[r];
        }
        const _Float16* pr = lp + ln * 40 + hf * 8;   // same-wave LDS is in-order
        const v16h pfrag = CAT16(*(const v8h*)pr, *(const v8h*)(pr + 16));

#pragma unroll
        for (int r = 0; r < 8; ++r) {
            yacc0[r] *= corr[r]; yacc1[r] *= corr[r];
            yacc2[r] *= corr[r]; yacc3[r] *= corr[r];
        }
        const _Float16* pv = xhb + (size_t)ln * NN + m0 + hf * 8;
        yacc0 = WMMA_F16(pfrag, load_bfrag(pv),                   yacc0);
        yacc1 = WMMA_F16(pfrag, load_bfrag(pv + (size_t)16 * NN), yacc1);
        yacc2 = WMMA_F16(pfrag, load_bfrag(pv + (size_t)32 * NN), yacc2);
        yacc3 = WMMA_F16(pfrag, load_bfrag(pv + (size_t)48 * NN), yacc3);
    }

    // ---------------- Finalize: single cross-lane sum, then write Y ----------------
    float inv[8];
#pragma unroll
    for (int r = 0; r < 8; ++r) inv[r] = 1.0f / half16_sum(psum[r]);

    {
        const v8f yv[4] = { yacc0, yacc1, yacc2, yacc3 };
        float* py = out_y + ((size_t)b * CC + ln) * NN + r0 + hf * 8;
#pragma unroll
        for (int ct = 0; ct < 4; ++ct) {
            v4f lo, hi;
#pragma unroll
            for (int r = 0; r < 4; ++r) {
                lo[r] = yv[ct][r]     * inv[r];
                hi[r] = yv[ct][r + 4] * inv[r + 4];
            }
            float* pc = py + (size_t)(ct * 16) * NN;
            *(v4f*)pc       = lo;
            *(v4f*)(pc + 4) = hi;
        }
    }

    // ---------------- Pass 2: recompute S, stream out normalized attn ----------------
    float* pattn = out_attn + (size_t)b * NN * NN + (size_t)(r0 + hf * 8) * NN + ln;
    for (int j = 0; j < NN / 16; ++j) {
        const int m0 = j << 4;
        v8f s = s_tile16(xTb, m0, ln, hf, a0, a1);
#pragma unroll
        for (int r = 0; r < 8; ++r) {
            const float p = fast_exp2(s[r] - rmax[r]) * inv[r];
            __builtin_nontemporal_store(p, pattn + (size_t)r * NN + m0);
        }
    }
}

// ---------------------------------------------------------------------------
extern "C" void kernel_launch(void* const* d_in, const int* in_sizes, int n_in,
                              void* d_out, int out_size, void* d_ws, size_t ws_size,
                              hipStream_t stream) {
    const float* x = (const float*)d_in[0];
    float* out_y    = (float*)d_out;
    float* out_attn = out_y + (size_t)BB * CC * NN;

    _Float16* xT = (_Float16*)d_ws;
    _Float16* xh = xT + (size_t)BB * NN * CC;

    convert_f16_kernel<<<dim3(BB * (NN / 64)), dim3(256), 0, stream>>>(x, xT, xh);
    spatial_attn_kernel<<<dim3((BB * NN / 16) / 4), dim3(128), 0, stream>>>(xT, xh, out_y, out_attn);
}
